// SwinTransformerBlock_11450382811952
// MI455X (gfx1250) — matrix operand order repaired and checked by reading, not compiled, not run on hardware
//
#include <hip/hip_runtime.h>

typedef _Float16 half_t;
typedef __attribute__((ext_vector_type(16))) _Float16 v16h;
typedef __attribute__((ext_vector_type(8)))  float    v8f;
typedef __attribute__((ext_vector_type(4)))  float    f32x4;

union Frag { v16h v; f32x4 f4[2]; };

__device__ __forceinline__ v8f wmma16(const Frag& a, const Frag& b, v8f c) {
  return __builtin_amdgcn_wmma_f32_16x16x32_f16(false, a.v, false, b.v,
                                                (short)0, c, false, false);
}

// ---- gfx1250 async global->LDS copy (ASYNCcnt-tracked, ISA 15.18.3) ----
__device__ __forceinline__ void async_b128(unsigned ldsoff, unsigned long long g) {
  asm volatile("global_load_async_to_lds_b128 %0, %1, off"
               :: "v"(ldsoff), "v"(g) : "memory");
}
__device__ __forceinline__ void wait_async0() {
  asm volatile("s_wait_asynccnt 0x0" ::: "memory");
}
__device__ __forceinline__ void wait_async2() {
  asm volatile("s_wait_asynccnt 0x2" ::: "memory");
}

// ---------------- constants ----------------
#define BQ   32
#define HH0  48
#define CC   384
#define NHH  12
#define WSZ  12
#define SHF  6
#define NN   144                    // tokens per window
#define NWIN 512                    // 32 * 16
#define MTOT (NWIN * NN)            // 73728 rows
#define QKVE ((size_t)NWIN * NHH * NN * 32)   // 28,311,552 elems per q/k/v

// ================= CPB MLP: tab[529][12] =================
__global__ __launch_bounds__(128)
void k_cpb(const float* __restrict__ table, const float* __restrict__ w1,
           const float* __restrict__ b1, const float* __restrict__ w2,
           float* __restrict__ tab) {
  __shared__ float hid[512];
  int t0 = blockIdx.x;                      // 0..528
  float x0 = table[t0 * 2 + 0];
  float x1 = table[t0 * 2 + 1];
  int t = threadIdx.x;
  for (int j = t; j < 512; j += 128)
    hid[j] = fmaxf(0.f, x0 * w1[j] + x1 * w1[512 + j] + b1[j]);
  __syncthreads();
  if (t < NHH) {
    float s = 0.f;
    for (int j = 0; j < 512; j++) s += hid[j] * w2[j * NHH + t];
    tab[t0 * NHH + t] = s;
  }
}

// ============ bias[h][n][m] = 16*sigmoid(tab[rpi[n][m]][h]) ============
__global__ __launch_bounds__(256)
void k_bias(const float* __restrict__ tab, const int* __restrict__ rpi,
            float* __restrict__ bias) {
  int idx = blockIdx.x * 256 + threadIdx.x;          // h*144*144 + n*144 + m
  if (idx >= NHH * NN * NN) return;
  int h = idx / (NN * NN);
  int rem = idx - h * NN * NN;
  float v = tab[rpi[rem] * NHH + h];
  bias[idx] = 16.f / (1.f + __expf(-v));
}

// ============ shift + window partition + fp32->f16 ============
__global__ __launch_bounds__(256)
void k_gather(const float* __restrict__ x, half_t* __restrict__ xw) {
  long long e = ((long long)blockIdx.x * 256 + threadIdx.x) * 4;
  if (e >= (long long)MTOT * CC) return;
  int r = (int)(e / CC), c = (int)(e % CC);
  int w = r / NN, n = r - w * NN;
  int b = w >> 4, wi = w & 15, wh = wi >> 2, wc = wi & 3;
  int i = n / WSZ, j = n - i * WSZ;
  int hh = wh * WSZ + i + SHF; if (hh >= HH0) hh -= HH0;
  int wp = wc * WSZ + j + SHF; if (wp >= HH0) wp -= HH0;
  f32x4 v = *(const f32x4*)(x + ((size_t)b * (HH0 * HH0) + hh * HH0 + wp) * CC + c);
  half_t* o = xw + e;
  o[0] = (half_t)v.x; o[1] = (half_t)v.y; o[2] = (half_t)v.z; o[3] = (half_t)v.w;
}

// ================= generic WMMA GEMM (async double-buffered) =================
// block tile 128x64, 8 waves, each wave 32x32 (2x2 WMMA tiles), K-step 32.
// A staged via global_load_async_to_lds_b128 (2 bufs); B reg-prefetched f32->f16.
// MODE 0: +qkv bias, scatter to Q/K/V [p][w][h][n][d]  (f16)
// MODE 1: +proj_b, window-reverse+roll scatter (f32)
// MODE 2: +fc1_b, exact GELU, row-major f16
// MODE 3: +fc2_b, row-major f32
template <int MODE>
__global__ __launch_bounds__(256)
void k_gemm(const half_t* __restrict__ A, const float* __restrict__ W,
            const float* __restrict__ b0, const float* __restrict__ b1,
            int K, int Nc, half_t* __restrict__ outH, float* __restrict__ outF) {
  __shared__ __align__(16) half_t As[2][128 * 32];
  __shared__ __align__(16) half_t Bs[2][64 * 32];   // transposed [n][k]
  int t = threadIdx.x;
  int rb = blockIdx.y * 128;
  int cb = blockIdx.x * 64;
  int wave = t >> 5, L = t & 31, l16 = L & 15, g = L >> 4;
  int wm = wave & 3, wn = wave >> 2;
  int arow = t >> 1, acol = (t & 1) * 16;
  const half_t* abase = A + (size_t)(rb + arow) * K + acol;
  // generic->LDS: low 32 bits of flat shared address are the LDS byte offset
  unsigned aoff[2] = { (unsigned)(size_t)(&As[0][arow * 32 + acol]),
                       (unsigned)(size_t)(&As[1][arow * 32 + acol]) };
  int niter = K / 32;
  v8f acc[2][2] = {};
  f32x4 breg[2];

  auto issueA = [&](int i) {
    unsigned lo = aoff[i & 1];
    unsigned long long gp = (unsigned long long)(size_t)(abase + (size_t)i * 32);
    async_b128(lo, gp);
    async_b128(lo + 16u, gp + 16ull);
  };
  auto loadB = [&](int i) {
#pragma unroll
    for (int j = 0; j < 2; j++) {
      int qq = t * 2 + j; int kk = qq >> 4; int nn = (qq & 15) * 4;
      breg[j] = *(const f32x4*)(W + (size_t)(i * 32 + kk) * Nc + cb + nn);
    }
  };
  auto storeB = [&](int i) {
    half_t* bs = Bs[i & 1];
#pragma unroll
    for (int j = 0; j < 2; j++) {
      int qq = t * 2 + j; int kk = qq >> 4; int nn = (qq & 15) * 4;
      bs[(nn + 0) * 32 + kk] = (half_t)breg[j].x;
      bs[(nn + 1) * 32 + kk] = (half_t)breg[j].y;
      bs[(nn + 2) * 32 + kk] = (half_t)breg[j].z;
      bs[(nn + 3) * 32 + kk] = (half_t)breg[j].w;
    }
  };
  auto computeT = [&](int i) {
    const half_t* as = As[i & 1];
    const half_t* bsx = Bs[i & 1];
#pragma unroll
    for (int sub = 0; sub < 2; sub++) {
      Frag fa;
      const half_t* ar = as + (wm * 32 + sub * 16 + l16) * 32;
      fa.f4[0] = *(const f32x4*)(ar + g * 8);
      fa.f4[1] = *(const f32x4*)(ar + 16 + g * 8);
#pragma unroll
      for (int tc = 0; tc < 2; tc++) {
        Frag fb;
        const half_t* br = bsx + (wn * 32 + tc * 16 + l16) * 32 + g * 8;
        fb.f4[0] = *(const f32x4*)(br);
        fb.f4[1] = *(const f32x4*)(br + 16);
        acc[sub][tc] = wmma16(fa, fb, acc[sub][tc]);
      }
    }
  };

  // prologue: tile 0 (async A + reg B), tile 1 async A in flight
  issueA(0);
  loadB(0);
  storeB(0);
  if (niter > 1) { issueA(1); wait_async2(); } else { wait_async0(); }
  __syncthreads();

  for (int i = 0; i < niter; i++) {
    if (i + 1 < niter) loadB(i + 1);       // overlap global B with WMMA
    computeT(i);
    __syncthreads();                        // all waves done with tile i
    if (i + 1 < niter) {
      storeB(i + 1);
      if (i + 2 < niter) { issueA(i + 2); wait_async2(); }
      else               { wait_async0(); }
      __syncthreads();                      // tile i+1 LDS ready everywhere
    }
  }

  // epilogue
#pragma unroll
  for (int sub = 0; sub < 2; sub++)
#pragma unroll
    for (int tc = 0; tc < 2; tc++)
#pragma unroll
      for (int r = 0; r < 8; r++) {
        int row = rb + wm * 32 + sub * 16 + r + g * 8;
        int col = cb + wn * 32 + tc * 16 + l16;
        float val = acc[sub][tc][r];
        if (MODE == 0) {
          val += (col < CC) ? b0[col] : (col < 2 * CC ? 0.f : b1[col - 2 * CC]);
          int w = row / NN, n = row - w * NN;
          int p = col / CC, cc = col - p * CC, hdI = cc >> 5, d = cc & 31;
          size_t dst = (size_t)p * QKVE + ((size_t)w * NHH + hdI) * (NN * 32) + n * 32 + d;
          outH[dst] = (half_t)val;
        } else if (MODE == 1) {
          val += b0[col];
          int w = row / NN, n = row - w * NN;
          int b_ = w >> 4, wi = w & 15, wh = wi >> 2, wc = wi & 3;
          int i = n / WSZ, j = n - i * WSZ;
          int hh = wh * WSZ + i + SHF; if (hh >= HH0) hh -= HH0;
          int wp = wc * WSZ + j + SHF; if (wp >= HH0) wp -= HH0;
          outF[((size_t)b_ * (HH0 * HH0) + hh * HH0 + wp) * CC + col] = val;
        } else if (MODE == 2) {
          val += b0[col];
          val = 0.5f * val * (1.f + erff(val * 0.70710678f));
          outH[(size_t)row * Nc + col] = (half_t)val;
        } else {
          val += b0[col];
          outF[(size_t)row * Nc + col] = val;
        }
      }
}

// ================= fused attention per (window, head) =================
// 9 waves. S = (qn @ kn^T)*scale + bias + mask ; softmax ; O = P @ V
__global__ __launch_bounds__(288)
void k_attn(const half_t* __restrict__ Q, const half_t* __restrict__ Km,
            const half_t* __restrict__ V, const float* __restrict__ bias,
            const float* __restrict__ mask, const float* __restrict__ lsc,
            half_t* __restrict__ actx) {
  __shared__ __align__(16) half_t qn[NN * 32];
  __shared__ __align__(16) half_t kn[NN * 32];
  __shared__ __align__(16) half_t vt[32 * 160];     // V^T, K-padded to 160
  __shared__ __align__(16) float  S[NN * NN];       // also holds P (f16, stride 288)
  int wid = blockIdx.x;
  int w = wid / NHH, h = wid - w * NHH;
  int t = threadIdx.x;
  size_t base = ((size_t)w * NHH + h) * (NN * 32);
  const half_t* qptr = Q + base;
  const half_t* kptr = Km + base;
  const half_t* vptr = V + base;

  // load + L2-normalize q and k rows (one row per thread: 144 + 144)
  {
    int row = (t < NN) ? t : t - NN;
    const half_t* src = ((t < NN) ? qptr : kptr) + row * 32;
    float tmp[32]; float ss = 0.f;
#pragma unroll
    for (int d = 0; d < 32; d++) { float f = (float)src[d]; tmp[d] = f; ss += f * f; }
    float inv = 1.f / fmaxf(sqrtf(ss), 1e-12f);
    half_t* dst = ((t < NN) ? qn : kn) + row * 32;
#pragma unroll
    for (int d = 0; d < 32; d++) dst[d] = (half_t)(tmp[d] * inv);
  }
  // V transposed into LDS with zero K-padding
  for (int idx = t; idx < NN * 32; idx += 288) {
    int m = idx >> 5, d = idx & 31;
    vt[d * 160 + m] = vptr[idx];
  }
  for (int idx = t; idx < 32 * 16; idx += 288) {
    int d = idx >> 4, m = NN + (idx & 15);
    vt[d * 160 + m] = (half_t)0.f;
  }
  __syncthreads();

  int ww = t >> 5, L = t & 31, l16 = L & 15, g = L >> 4;

  // ---- S = qn @ kn^T : wave ww owns rows [16ww,16ww+16), 9 col tiles ----
  {
    Frag fa;
    const half_t* ar = qn + (16 * ww + l16) * 32;
    fa.f4[0] = *(const f32x4*)(ar + g * 8);
    fa.f4[1] = *(const f32x4*)(ar + 16 + g * 8);
    v8f acc[9] = {};
#pragma unroll
    for (int tc = 0; tc < 9; tc++) {
      Frag fb;
      const half_t* br = kn + (tc * 16 + l16) * 32 + g * 8;
      fb.f4[0] = *(const f32x4*)(br);
      fb.f4[1] = *(const f32x4*)(br + 16);
      acc[tc] = wmma16(fa, fb, acc[tc]);
    }
    float scale = __expf(fminf(lsc[h], 4.6051702f));
    int widx = w & 15;
#pragma unroll
    for (int tc = 0; tc < 9; tc++)
#pragma unroll
      for (int r = 0; r < 8; r++) {
        int n = 16 * ww + r + g * 8;
        int m = tc * 16 + l16;
        float v = acc[tc][r] * scale
                + bias[((size_t)h * NN + n) * NN + m]
                + mask[((size_t)widx * NN + n) * NN + m];
        S[n * NN + m] = v;
      }
  }
  __syncthreads();

  // ---- row softmax; write P (f16) in place, stride 288 halfs, pad K to 160 ----
  if (t < NN) {
    float* srow = S + t * NN;
    float mx = -1e30f;
    for (int m = 0; m < NN; m++) mx = fmaxf(mx, srow[m]);
    float sum = 0.f;
    for (int m = 0; m < NN; m++) sum += __expf(srow[m] - mx);
    float inv = 1.f / sum;
    volatile float* vs = (volatile float*)srow;
    volatile half_t* prow = (volatile half_t*)((half_t*)S + t * 288);
    for (int m = 0; m < NN; m++) {
      float e = __expf(vs[m] - mx) * inv;   // read dword m, write dword m/2: safe
      prow[m] = (half_t)e;
    }
    for (int m = NN; m < 160; m++) prow[m] = (half_t)0.f;
  }
  __syncthreads();

  // ---- O = P @ V : wave ww rows, 2 col tiles (d 0..31), K=160 in 5 steps ----
  {
    v8f oacc[2] = {};
    const half_t* prow = (const half_t*)S + (16 * ww + l16) * 288;
#pragma unroll
    for (int kb = 0; kb < 160; kb += 32) {
      Frag fa;
      fa.f4[0] = *(const f32x4*)(prow + kb + g * 8);
      fa.f4[1] = *(const f32x4*)(prow + kb + 16 + g * 8);
#pragma unroll
      for (int nb = 0; nb < 2; nb++) {
        Frag fb;
        const half_t* vr = vt + (nb * 16 + l16) * 160 + kb + g * 8;
        fb.f4[0] = *(const f32x4*)(vr);
        fb.f4[1] = *(const f32x4*)(vr + 16);
        oacc[nb] = wmma16(fa, fb, oacc[nb]);
      }
    }
#pragma unroll
    for (int nb = 0; nb < 2; nb++)
#pragma unroll
      for (int r = 0; r < 8; r++) {
        int grow = w * NN + 16 * ww + r + g * 8;
        int col = h * 32 + nb * 16 + l16;
        actx[(size_t)grow * CC + col] = (half_t)oacc[nb][r];
      }
  }
}

// ================= LayerNorm + residual =================
__global__ __launch_bounds__(128)
void k_ln1(const float* __restrict__ y, const float* __restrict__ x0,
           const float* __restrict__ gg, const float* __restrict__ bb,
           float* __restrict__ x1, half_t* __restrict__ x1h) {
  __shared__ float red[128];
  int r = blockIdx.x, t = threadIdx.x;
  const float* yr = y + (size_t)r * CC;
  float v0 = yr[t], v1 = yr[t + 128], v2 = yr[t + 256];
  red[t] = v0 + v1 + v2; __syncthreads();
  for (int o = 64; o > 0; o >>= 1) { if (t < o) red[t] += red[t + o]; __syncthreads(); }
  float mean = red[0] * (1.f / CC); __syncthreads();
  red[t] = v0 * v0 + v1 * v1 + v2 * v2; __syncthreads();
  for (int o = 64; o > 0; o >>= 1) { if (t < o) red[t] += red[t + o]; __syncthreads(); }
  float var = red[0] * (1.f / CC) - mean * mean;
  float inv = rsqrtf(var + 1e-5f);
  const float* xr = x0 + (size_t)r * CC;
#pragma unroll
  for (int i = 0; i < 3; i++) {
    int c = t + i * 128;
    float v = (i == 0) ? v0 : (i == 1) ? v1 : v2;
    float o = xr[c] + (v - mean) * inv * gg[c] + bb[c];
    x1[(size_t)r * CC + c] = o;
    x1h[(size_t)r * CC + c] = (half_t)o;
  }
}

__global__ __launch_bounds__(128)
void k_ln2(const float* __restrict__ y, const float* __restrict__ x1,
           const float* __restrict__ gg, const float* __restrict__ bb,
           float* __restrict__ out) {
  __shared__ float red[128];
  int r = blockIdx.x, t = threadIdx.x;
  const float* yr = y + (size_t)r * CC;
  float v0 = yr[t], v1 = yr[t + 128], v2 = yr[t + 256];
  red[t] = v0 + v1 + v2; __syncthreads();
  for (int o = 64; o > 0; o >>= 1) { if (t < o) red[t] += red[t + o]; __syncthreads(); }
  float mean = red[0] * (1.f / CC); __syncthreads();
  red[t] = v0 * v0 + v1 * v1 + v2 * v2; __syncthreads();
  for (int o = 64; o > 0; o >>= 1) { if (t < o) red[t] += red[t + o]; __syncthreads(); }
  float var = red[0] * (1.f / CC) - mean * mean;
  float inv = rsqrtf(var + 1e-5f);
  const float* xr = x1 + (size_t)r * CC;
#pragma unroll
  for (int i = 0; i < 3; i++) {
    int c = t + i * 128;
    float v = (i == 0) ? v0 : (i == 1) ? v1 : v2;
    out[(size_t)r * CC + c] = xr[c] + (v - mean) * inv * gg[c] + bb[c];
  }
}

// ================= launch =================
extern "C" void kernel_launch(void* const* d_in, const int* in_sizes, int n_in,
                              void* d_out, int out_size, void* d_ws, size_t ws_size,
                              hipStream_t stream) {
  const float* x        = (const float*)d_in[0];
  const float* rct      = (const float*)d_in[1];
  const int*   rpi      = (const int*)d_in[2];
  const float* amask    = (const float*)d_in[3];
  const float* qkv_w    = (const float*)d_in[4];
  const float* q_bias   = (const float*)d_in[5];
  const float* v_bias   = (const float*)d_in[6];
  const float* lsc      = (const float*)d_in[7];
  const float* cpb_w1   = (const float*)d_in[8];
  const float* cpb_b1   = (const float*)d_in[9];
  const float* cpb_w2   = (const float*)d_in[10];
  const float* proj_w   = (const float*)d_in[11];
  const float* proj_b   = (const float*)d_in[12];
  const float* n1g      = (const float*)d_in[13];
  const float* n1b      = (const float*)d_in[14];
  const float* n2g      = (const float*)d_in[15];
  const float* n2b      = (const float*)d_in[16];
  const float* fc1_w    = (const float*)d_in[17];
  const float* fc1_b    = (const float*)d_in[18];
  const float* fc2_w    = (const float*)d_in[19];
  const float* fc2_b    = (const float*)d_in[20];

  const size_t MB = 1024ull * 1024ull;
  char* w8 = (char*)d_ws;
  float*  tabf  = (float*)(w8 + 0);            // 529*12 f32
  float*  biasf = (float*)(w8 + (32 << 10));   // 12*144*144 f32 (~0.95 MiB)
  half_t* xw    = (half_t*)(w8 + 2 * MB);      // 54 MiB f16
  half_t* qkvh  = (half_t*)(w8 + 56 * MB);     // 3 * 54 MiB f16
  half_t* Qh = qkvh;
  half_t* Kh = qkvh + QKVE;
  half_t* Vh = Kh + QKVE;
  half_t* actx  = (half_t*)(w8 + 218 * MB);    // 54 MiB f16
  float*  y4    = (float*)(w8 + 2 * MB);       // reuse xw/Q (dead)  108 MiB f32
  float*  x1    = (float*)(w8 + 110 * MB);     // reuse K/V (dead)   108 MiB f32
  half_t* x1h   = (half_t*)(w8 + 218 * MB);    // reuse actx (dead)  54 MiB f16
  half_t* hid   = (half_t*)(w8 + 272 * MB);    // 216 MiB f16
  float*  h2    = (float*)(w8 + 2 * MB);       // reuse y4 (dead)    108 MiB f32
  float*  out   = (float*)d_out;

  k_cpb   <<<529, 128, 0, stream>>>(rct, cpb_w1, cpb_b1, cpb_w2, tabf);
  k_bias  <<<(NHH * NN * NN) / 256, 256, 0, stream>>>(tabf, rpi, biasf);
  k_gather<<<(int)(((size_t)MTOT * CC) / 1024), 256, 0, stream>>>(x, xw);
  k_gemm<0><<<dim3(1152 / 64, MTOT / 128), 256, 0, stream>>>(
      xw, qkv_w, q_bias, v_bias, 384, 1152, qkvh, nullptr);
  k_attn  <<<NWIN * NHH, 288, 0, stream>>>(Qh, Kh, Vh, biasf, amask, lsc, actx);
  k_gemm<1><<<dim3(384 / 64, MTOT / 128), 256, 0, stream>>>(
      actx, proj_w, proj_b, nullptr, 384, 384, nullptr, y4);
  k_ln1   <<<MTOT, 128, 0, stream>>>(y4, x, n1g, n1b, x1, x1h);
  k_gemm<2><<<dim3(1536 / 64, MTOT / 128), 256, 0, stream>>>(
      x1h, fc1_w, fc1_b, nullptr, 384, 1536, hid, nullptr);
  k_gemm<3><<<dim3(384 / 64, MTOT / 128), 256, 0, stream>>>(
      hid, fc2_w, fc2_b, nullptr, 1536, 384, nullptr, h2);
  k_ln2   <<<MTOT, 128, 0, stream>>>(h2, x1, n2g, n2b, out);
}
